// Attention_25890062860783
// MI455X (gfx1250) — matrix operand order repaired and checked by reading, not compiled, not run on hardware
//
#include <hip/hip_runtime.h>
#include <hip/hip_bf16.h>

// ---------------------------------------------------------------------------
// B=256, S=2048, F=256, d=64.
//   q,k,v = ReLU(LN(x@W + b))                  (WMMA bf16, f32 accum)
//   att_s = (Q_s K_s^T) V_s == Q_s (K_s^T V_s) (no softmax -> associate:
//                                               inner product is only 64x64)
// k1 tiles rows as (fixed s, 16 consecutive b) so the transposed kT/vT
// stores become contiguous 16B global_store_b128 per lane.
// ---------------------------------------------------------------------------

typedef __bf16 bf16_t;
typedef __attribute__((ext_vector_type(16))) __bf16 v16bf;
typedef __attribute__((ext_vector_type(8)))  __bf16 v8bf;
typedef __attribute__((ext_vector_type(8)))  float  v8f;
typedef __attribute__((ext_vector_type(4)))  float  v4f;

#define S_LEN 2048
#define B_LEN 256
#define F_DIM 256
#define D_DIM 64

union V16U { v16bf v; v8bf h[2]; };

static __device__ __forceinline__ v16bf combine(v8bf lo, v8bf hi) {
  V16U u; u.h[0] = lo; u.h[1] = hi; return u.v;
}

static __device__ __forceinline__ v8f zero8() {
  v8f z = {0.f, 0.f, 0.f, 0.f, 0.f, 0.f, 0.f, 0.f};
  return z;
}

static __device__ __forceinline__ v8f wmma_bf16(v16bf a, v16bf b, v8f c) {
  return __builtin_amdgcn_wmma_f32_16x16x32_bf16(false, a, false, b, (short)0, c,
                                                 false, false);
}

// ---------------------------------------------------------------------------
// k0: repack Wq/Wk/Wv (f32 [256][64]) -> WT bf16 [3][64][256] (transposed)
// ---------------------------------------------------------------------------
__global__ void qkv_prep_kernel(const float* __restrict__ Wq,
                                const float* __restrict__ Wk,
                                const float* __restrict__ Wv,
                                bf16_t* __restrict__ WT) {
  const int total = 3 * D_DIM * F_DIM;
  for (int i = threadIdx.x + blockIdx.x * blockDim.x; i < total;
       i += blockDim.x * gridDim.x) {
    int p = i / (D_DIM * F_DIM);
    int r = i - p * (D_DIM * F_DIM);
    int n = r / F_DIM;        // output feature 0..63
    int k = r - n * F_DIM;    // input  feature 0..255
    const float* W = (p == 0) ? Wq : ((p == 1) ? Wk : Wv);
    WT[i] = (bf16_t)W[k * D_DIM + n];
  }
}

// ---------------------------------------------------------------------------
// k1: fused QKV projection + bias + LayerNorm + ReLU.
// Wave tile = 16 consecutive b rows at one fixed s (tile id g = blk*8+wave,
// b0 = (g&15)*16, s = g>>4). Shared A fragment drives 12 accumulators
// (3 proj x 4 n-tiles); rotating B register + prefetched next-A raw floats
// keep global loads overlapped with the WMMA chain.
// Outputs: q [s][b][d]; kT,vT [s][d][b] (packed 16B stores along b).
// ---------------------------------------------------------------------------
__global__ void __launch_bounds__(256)
qkv_proj_kernel(const float* __restrict__ x,
                const bf16_t* __restrict__ WT,
                const float* __restrict__ bq, const float* __restrict__ gq,
                const float* __restrict__ beq,
                const float* __restrict__ bk, const float* __restrict__ gk,
                const float* __restrict__ bek,
                const float* __restrict__ bv, const float* __restrict__ gv,
                const float* __restrict__ bev,
                bf16_t* __restrict__ q, bf16_t* __restrict__ kT,
                bf16_t* __restrict__ vT) {
  const int lane = threadIdx.x & 31;
  const int wave = threadIdx.x >> 5;
  const int l15  = lane & 15;
  const int hi   = (lane >> 4) & 1;

  const long g  = (long)blockIdx.x * 8 + wave;   // 0..32767
  const int  b0 = (int)(g & 15) * 16;            // batch tile
  const long ss = g >> 4;                        // seq position

  // A rows: b = b0 + l15 (fixed s); contiguous along input features.
  const float* xrow =
      x + ((size_t)(b0 + l15) * S_LEN + ss) * F_DIM + hi * 8;
  // B rows: WT[(p*64 + t*16 + l15)][k], fragment offset = pt*16 rows.
  const bf16_t* wbase = WT + (size_t)l15 * F_DIM + hi * 8;

  v8f acc[12];
#pragma unroll
  for (int i = 0; i < 12; ++i) acc[i] = zero8();

  // preload A raw floats for c=0 (non-temporal: x is streamed exactly once)
  v4f ra0 = __builtin_nontemporal_load((const v4f*)(xrow));
  v4f ra1 = __builtin_nontemporal_load((const v4f*)(xrow + 4));
  v4f ra2 = __builtin_nontemporal_load((const v4f*)(xrow + 16));
  v4f ra3 = __builtin_nontemporal_load((const v4f*)(xrow + 20));

#pragma unroll
  for (int c = 0; c < F_DIM; c += 32) {
    v16bf A;
#pragma unroll
    for (int j = 0; j < 4; ++j) {
      A[j]      = (bf16_t)ra0[j];
      A[4 + j]  = (bf16_t)ra1[j];
      A[8 + j]  = (bf16_t)ra2[j];
      A[12 + j] = (bf16_t)ra3[j];
    }
    if (c + 32 < F_DIM) {  // prefetch next A tile while WMMAs run
      ra0 = __builtin_nontemporal_load((const v4f*)(xrow + c + 32));
      ra1 = __builtin_nontemporal_load((const v4f*)(xrow + c + 36));
      ra2 = __builtin_nontemporal_load((const v4f*)(xrow + c + 48));
      ra3 = __builtin_nontemporal_load((const v4f*)(xrow + c + 52));
    }
    const bf16_t* bp = wbase + c;
    v16bf B = combine(*(const v8bf*)bp, *(const v8bf*)(bp + 16));
#pragma unroll
    for (int idx = 0; idx < 12; ++idx) {  // idx = p*4 + t
      v16bf Bn;
      if (idx < 11) {
        const bf16_t* bn = bp + (size_t)(idx + 1) * 16 * F_DIM;
        Bn = combine(*(const v8bf*)bn, *(const v8bf*)(bn + 16));
      }
      acc[idx] = wmma_bf16(A, B, acc[idx]);
      if (idx < 11) B = Bn;
    }
  }

  const float* bias_p[3] = {bq, bk, bv};
  const float* g_p[3]    = {gq, gk, gv};
  const float* be_p[3]   = {beq, bek, bev};

#pragma unroll
  for (int p = 0; p < 3; ++p) {
    float bias_r[4], g_r[4], be_r[4];
#pragma unroll
    for (int t = 0; t < 4; ++t) {
      const int d = t * 16 + l15;
      bias_r[t] = bias_p[p][d];
      g_r[t]    = g_p[p][d];
      be_r[t]   = be_p[p][d];
    }
#pragma unroll
    for (int t = 0; t < 4; ++t)
#pragma unroll
      for (int i = 0; i < 8; ++i) acc[p * 4 + t][i] += bias_r[t];

    v8bf pk[4];  // packed transposed outputs (kT/vT): 8 consecutive b

#pragma unroll
    for (int i = 0; i < 8; ++i) {  // VGPR slot i -> b row i (lo) / i+8 (hi)
      float sm = 0.f, sq = 0.f;
#pragma unroll
      for (int t = 0; t < 4; ++t) {
        const float h = acc[p * 4 + t][i];
        sm += h;
        sq += h * h;
      }
      // butterfly within the 16-lane half that owns this row
      sm += __shfl_xor(sm, 1, 32);  sq += __shfl_xor(sq, 1, 32);
      sm += __shfl_xor(sm, 2, 32);  sq += __shfl_xor(sq, 2, 32);
      sm += __shfl_xor(sm, 4, 32);  sq += __shfl_xor(sq, 4, 32);
      sm += __shfl_xor(sm, 8, 32);  sq += __shfl_xor(sq, 8, 32);
      const float mu = sm * (1.0f / 64.0f);
      const float rs = rsqrtf(sq * (1.0f / 64.0f) - mu * mu + 1e-5f);

      const long bb = b0 + i + hi * 8;
#pragma unroll
      for (int t = 0; t < 4; ++t) {
        const int d = t * 16 + l15;
        float y = (acc[p * 4 + t][i] - mu) * rs * g_r[t] + be_r[t];
        y = fmaxf(y, 0.0f);
        const bf16_t yv = (bf16_t)y;
        if (p == 0) q[(ss * B_LEN + bb) * D_DIM + d] = yv;
        else        pk[t][i] = yv;
      }
    }

    if (p > 0) {  // kT/vT: one 16B store per tile (8 consecutive b)
      bf16_t* dst = (p == 1) ? kT : vT;
#pragma unroll
      for (int t = 0; t < 4; ++t) {
        bf16_t* addr = dst + (ss * D_DIM + t * 16 + l15) * B_LEN + b0 + hi * 8;
        *(v8bf*)addr = pk[t];
      }
    }
  }
}

// ---------------------------------------------------------------------------
// k2: one workgroup (16 waves) per seq position s.
//   Phase A: M = K^T V (64x64): 16 WMMA tiles (1/wave), K-loop over b=256;
//            M stored transposed (MT[dv][dk]) bf16 in 8KB LDS via one
//            ds_store_b128 per lane.
//   Phase B: att = Q @ M: 16 b-tiles (1/wave) x 4 d-tiles, 2 WMMA each;
//            non-temporal f32 output stores.
// ---------------------------------------------------------------------------
__global__ void __launch_bounds__(512)
attn_kernel(const bf16_t* __restrict__ q, const bf16_t* __restrict__ kT,
            const bf16_t* __restrict__ vT, float* __restrict__ out) {
  __shared__ bf16_t MT[D_DIM * D_DIM];  // MT[dv][dk]

  const int lane = threadIdx.x & 31;
  const int wave = threadIdx.x >> 5;
  const int l15  = lane & 15;
  const int hi   = (lane >> 4) & 1;
  const long s   = blockIdx.x;

  // ---- Phase A: M = K^T V -------------------------------------------------
  {
    const int m0 = (wave >> 2) * 16;  // dk tile
    const int n0 = (wave & 3) * 16;   // dv tile
    const bf16_t* arow = kT + (s * D_DIM + m0 + l15) * B_LEN;
    const bf16_t* brow = vT + (s * D_DIM + n0 + l15) * B_LEN;
    v8f acc = zero8();
#pragma unroll
    for (int c = 0; c < B_LEN; c += 32) {
      v16bf A  = combine(*(const v8bf*)(arow + c + hi * 8),
                         *(const v8bf*)(arow + c + 16 + hi * 8));
      v16bf Bf = combine(*(const v8bf*)(brow + c + hi * 8),
                         *(const v8bf*)(brow + c + 16 + hi * 8));
      acc = wmma_bf16(A, Bf, acc);
    }
    v8bf pk;
#pragma unroll
    for (int i = 0; i < 8; ++i) pk[i] = (bf16_t)acc[i];
    *(v8bf*)(&MT[(n0 + l15) * D_DIM + m0 + hi * 8]) = pk;  // ds_store_b128
  }
  __syncthreads();

  // ---- Phase B: att = Q @ M ----------------------------------------------
  {
    const int b0 = wave * 16;
    const bf16_t* qrow = q + (s * B_LEN + b0 + l15) * D_DIM;
    v16bf A0 = combine(*(const v8bf*)(qrow + hi * 8),
                       *(const v8bf*)(qrow + 16 + hi * 8));
    v16bf A1 = combine(*(const v8bf*)(qrow + 32 + hi * 8),
                       *(const v8bf*)(qrow + 48 + hi * 8));
#pragma unroll
    for (int nt = 0; nt < 4; ++nt) {
      const bf16_t* mrow = MT + (nt * 16 + l15) * D_DIM;
      v16bf B0 = combine(*(const v8bf*)(mrow + hi * 8),
                         *(const v8bf*)(mrow + 16 + hi * 8));
      v16bf B1 = combine(*(const v8bf*)(mrow + 32 + hi * 8),
                         *(const v8bf*)(mrow + 48 + hi * 8));
      v8f acc = zero8();
      acc = wmma_bf16(A0, B0, acc);
      acc = wmma_bf16(A1, B1, acc);
#pragma unroll
      for (int i = 0; i < 8; ++i) {
        const long bb = b0 + i + hi * 8;
        const int  d  = nt * 16 + l15;
        __builtin_nontemporal_store(acc[i],
                                    &out[(bb * S_LEN + s) * D_DIM + d]);
      }
    }
  }
}

// ---------------------------------------------------------------------------
extern "C" void kernel_launch(void* const* d_in, const int* in_sizes, int n_in,
                              void* d_out, int out_size, void* d_ws,
                              size_t ws_size, hipStream_t stream) {
  (void)in_sizes; (void)n_in; (void)out_size; (void)ws_size;

  const float* x   = (const float*)d_in[0];
  const float* Wq  = (const float*)d_in[1];
  const float* bq  = (const float*)d_in[2];
  const float* gq  = (const float*)d_in[3];
  const float* beq = (const float*)d_in[4];
  const float* Wk  = (const float*)d_in[5];
  const float* bk  = (const float*)d_in[6];
  const float* gk  = (const float*)d_in[7];
  const float* bek = (const float*)d_in[8];
  const float* Wv  = (const float*)d_in[9];
  const float* bv  = (const float*)d_in[10];
  const float* gv  = (const float*)d_in[11];
  const float* bev = (const float*)d_in[12];

  char* ws = (char*)d_ws;
  const size_t qkvBytes = (size_t)S_LEN * B_LEN * D_DIM * sizeof(bf16_t); // 64MB
  bf16_t* qb  = (bf16_t*)(ws);
  bf16_t* kTb = (bf16_t*)(ws + qkvBytes);
  bf16_t* vTb = (bf16_t*)(ws + 2 * qkvBytes);
  bf16_t* WT  = (bf16_t*)(ws + 3 * qkvBytes);

  qkv_prep_kernel<<<48, 256, 0, stream>>>(Wq, Wk, Wv, WT);

  // 2048 s * 16 b-tiles = 32768 wave tiles / 8 waves = 4096 blocks
  qkv_proj_kernel<<<4096, 256, 0, stream>>>(x, WT, bq, gq, beq, bk, gk, bek,
                                            bv, gv, bev, qb, kTb, vTb);

  attn_kernel<<<S_LEN, 512, 0, stream>>>(qb, kTb, vTb, (float*)d_out);
}
